// Linear2Bit_14980845928669
// MI455X (gfx1250) — matrix operand order repaired and checked by reading, not compiled, not run on hardware
//
#include <hip/hip_runtime.h>
#include <hip/hip_fp16.h>

typedef __attribute__((ext_vector_type(16))) _Float16 v16h;
typedef __attribute__((ext_vector_type(8)))  _Float16 v8h;
typedef __attribute__((ext_vector_type(8)))  float    v8f;

#define OUT_F 11008
#define IN_F  4096
#define N_TILES (OUT_F / 16)        // 688
#define WAVES_PER_BLOCK 8
#define SPLIT_K 8                   // K chunks of 512 -> 16 WMMA steps/wave
#define KSTEPS_PER_CHUNK (IN_F / 32 / SPLIT_K)   // 16

// ---------------------------------------------------------------------------
// Pre-kernel 1: convert x (f32, 32x4096) to f16 so A-fragment loads are
// single b128 loads of 8 contiguous f16.
// ---------------------------------------------------------------------------
__global__ void cvt_x_to_f16(const float* __restrict__ x,
                             _Float16* __restrict__ xh, int n) {
    int i = blockIdx.x * blockDim.x + threadIdx.x;
    if (i < n) xh[i] = (_Float16)x[i];
}

// ---------------------------------------------------------------------------
// Pre-kernel 2: out[m][o] = bias[o].  Makes kernel_launch idempotent: the
// split-K waves then accumulate partial sums with fp32 atomics.
// ---------------------------------------------------------------------------
__global__ void init_out_bias(float* __restrict__ out,
                              const float* __restrict__ bias, int n) {
    int i = blockIdx.x * blockDim.x + threadIdx.x;
    if (i < n) out[i] = bias[i % OUT_F];
}

// ---------------------------------------------------------------------------
// Fused 2-bit dequant + skinny GEMM using v_wmma_f32_16x16x32_f16.
// One wave owns one 16-wide N tile, both 16-row M tiles, and one K chunk.
// blockIdx.x = N-tile group (8 tiles), blockIdx.y = K chunk (shared by the
// block's 8 waves so their A-fragment reads hit WGP$).
//
// Dequant trick: 0x3C00 | (q<<8) is the f16 value (1 + q/4) exactly, so a
// packed half2 pair is built with pure logic ops and finished with one
// v_pk_fma_f16 using sc2 = 4*s, of2 = min - 4*s  =>  w = q*s + min.
// ---------------------------------------------------------------------------
__global__ __launch_bounds__(256) void linear2bit_wmma(
    const _Float16* __restrict__ xh,       // 32 x 4096 f16
    const int*      __restrict__ wp,       // (G, 32) packed int32, 4 codes/word
    const __half2*  __restrict__ ranges,   // (G,) {min, max} f16 pairs
    float*          __restrict__ out)      // 32 x 11008 f32 (pre-set to bias)
{
    const int lane = threadIdx.x & 31;
    const int wave = threadIdx.x >> 5;
    const int tile = blockIdx.x * WAVES_PER_BLOCK + wave;   // 0..687
    const int kc   = blockIdx.y;                            // K chunk 0..7
    const int o0   = tile * 16;
    const int ln   = lane & 15;     // B column / A row within tile
    const int hi   = lane >> 4;     // K-half selector

    const int o    = o0 + ln;              // this lane's output column
    const int grow = o * (IN_F / 128);     // first quant group of row o

    // A-fragment row base pointers (contiguous K: lo lanes K+0..15, hi lanes K+16..31)
    const _Float16* a0p = xh + (ln)      * IN_F + hi * 16;
    const _Float16* a1p = xh + (16 + ln) * IN_F + hi * 16;

    v8f acc0 = {}; v8f acc1 = {};
    __half2 sc2 = __float2half2_rn(0.f);
    __half2 of2 = __float2half2_rn(0.f);

    const int kk_beg = kc * KSTEPS_PER_CHUNK;      // multiple of 4
    const int kk_end = kk_beg + KSTEPS_PER_CHUNK;

    // B-stream base for this lane (advances 32B per K step).
    const int pos0 = (kk_beg * 32 & 127) + hi * 16;
    const int* pw  = wp + (grow + (kk_beg >> 2)) * 32 + (pos0 >> 2);

#pragma unroll 4
    for (int kk = kk_beg; kk < kk_end; ++kk) {
        const int k0 = kk * 32;

        // Per-group scale/offset: group spans 128 K -> changes every 4 steps.
        if ((kk & 3) == 0) {
            const int g = grow + (kk >> 2);
            __half2 r2 = ranges[g];
            const float rmin = __low2float(r2);
            const float rmax = __high2float(r2);
            const float s4 = (rmax - rmin) * (4.0f / 3.0f);
            sc2 = __float2half2_rn(s4);
            of2 = __float2half2_rn(rmin - s4);
        }

        // ---- B packed word load: 16 consecutive K of row o = one b128 ----
        // (B order used here, contiguous K as for A: lo lanes hold K 0..15, hi lanes K 16..31.)
        uint4 w = *(const uint4*)pw;
        __builtin_prefetch(pw + 64, 0, 0);           // global_prefetch_b8 ahead
        pw += 8;                                     // next K step (32B)

        // ---- A fragments: two b128 loads per M tile ----
        v16h a0, a1;
        {
            v8h l0 = *(const v8h*)(a0p + k0);
            v8h h0 = *(const v8h*)(a0p + k0 + 8);
            v8h l1 = *(const v8h*)(a1p + k0);
            v8h h1 = *(const v8h*)(a1p + k0 + 8);
#pragma unroll
            for (int i = 0; i < 8; ++i) {
                a0[i] = l0[i]; a0[8 + i] = h0[i];
                a1[i] = l1[i]; a1[8 + i] = h1[i];
            }
        }

        unsigned int wv[4] = {w.x, w.y, w.z, w.w};
        union { __half2 h2[8]; v16h v; } bu;
#pragma unroll
        for (int c = 0; c < 4; ++c) {
            unsigned int v = wv[c];
            __builtin_assume(v <= 255u);         // codes live in bits 0..7
            const unsigned int lo4 = v & 0xFu;          // q0 | q1<<2
            const unsigned int hi4 = (v >> 4) & 0xFu;   // q2 | q3<<2 (u24 mul)
            // spread q -> f16 mantissa bits 8-9 of each half: {1+q1/4, 1+q0/4}
            unsigned int M0 = ((lo4 * 0x00400100u) & 0x03000300u) | 0x3C003C00u;
            unsigned int M1 = ((hi4 * 0x00400100u) & 0x03000300u) | 0x3C003C00u;
            bu.h2[2 * c]     = __hfma2(__builtin_bit_cast(__half2, M0), sc2, of2);
            bu.h2[2 * c + 1] = __hfma2(__builtin_bit_cast(__half2, M1), sc2, of2);
        }
        v16h b = bu.v;

        // ---- WMMA: D = A x B + C (f32 accumulate), B reused for both tiles --
        acc0 = __builtin_amdgcn_wmma_f32_16x16x32_f16(
            false, a0, false, b, (short)0, acc0, false, false);
        acc1 = __builtin_amdgcn_wmma_f32_16x16x32_f16(
            false, a1, false, b, (short)0, acc1, false, false);
    }

    // ---- Epilogue: C/D layout lane = n + 16h, VGPR r -> M = r + 8h.
    // Split-K partials combined with hardware fp32 atomics (L2-resident).
#pragma unroll
    for (int r = 0; r < 8; ++r) {
        const int m0 = r + hi * 8;
        unsafeAtomicAdd(&out[(m0)      * OUT_F + o], acc0[r]);
        unsafeAtomicAdd(&out[(m0 + 16) * OUT_F + o], acc1[r]);
    }
}

// ---------------------------------------------------------------------------
extern "C" void kernel_launch(void* const* d_in, const int* in_sizes, int n_in,
                              void* d_out, int out_size, void* d_ws, size_t ws_size,
                              hipStream_t stream) {
    const float*   x      = (const float*)d_in[0];
    const int*     wp     = (const int*)d_in[1];
    const __half2* ranges = (const __half2*)d_in[2];
    const float*   bias   = (const float*)d_in[3];
    float*         out    = (float*)d_out;

    // Workspace: 32*4096 f16 = 256 KB for the converted activations.
    _Float16* xh = (_Float16*)d_ws;

    const int nx = 32 * IN_F;
    cvt_x_to_f16<<<(nx + 255) / 256, 256, 0, stream>>>(x, xh, nx);

    const int no = 32 * OUT_F;
    init_out_bias<<<(no + 255) / 256, 256, 0, stream>>>(out, bias, no);

    dim3 grid(N_TILES / WAVES_PER_BLOCK, SPLIT_K);   // 86 x 8 = 688 blocks
    linear2bit_wmma<<<grid, 32 * WAVES_PER_BLOCK, 0, stream>>>(
        xh, wp, ranges, out);
}
